// VGG16ROIHead_953482740203
// MI455X (gfx1250) — compile-verified, matrix operand order checked by reading, not run on hardware
//
#include <hip/hip_runtime.h>
#include <hip/hip_fp16.h>

typedef __attribute__((ext_vector_type(16))) _Float16 v16h;
typedef __attribute__((ext_vector_type(8)))  _Float16 v8h;
typedef __attribute__((ext_vector_type(4)))  _Float16 v4h;
typedef __attribute__((ext_vector_type(8)))  float    v8f;

#define OUTP 7
#define RR   128          // number of ROIs (== GEMM M)
#define CC   512
#define HH   37
#define WW   50
#define DIN  (CC*OUTP*OUTP)   // 25088
#define DH   4096
#define NSPLIT 8

#define BM 128
#define BN 128
#define BK 32
#define LDP 48            // padded LDS row stride in halves (96 B, keeps 16B alignment)

// ---------------------------------------------------------------------------
// 1) RoI max pooling: x (1,512,37,50) f32 -> A0 (128, 25088) f16
// ---------------------------------------------------------------------------
__global__ void roi_pool_kernel(const float* __restrict__ x,
                                const float* __restrict__ rois,
                                const int*   __restrict__ roi_idx,
                                _Float16*    __restrict__ outA)
{
    int idx = blockIdx.x * blockDim.x + threadIdx.x;
    const int total = RR * CC * OUTP * OUTP;
    if (idx >= total) return;
    int r   = idx / (CC * OUTP * OUTP);
    int rem = idx % (CC * OUTP * OUTP);
    int c   = rem / (OUTP * OUTP);
    int ij  = rem % (OUTP * OUTP);
    int i = ij / OUTP, j = ij % OUTP;

    float4 rv = ((const float4*)rois)[r];          // y1,x1,y2,x2
    int y1 = (int)floorf(rv.x * 0.0625f);
    int x1 = (int)floorf(rv.y * 0.0625f);
    int y2 = (int)floorf(rv.z * 0.0625f);
    int x2 = (int)floorf(rv.w * 0.0625f);
    int h = y2 - y1 + 1;
    int w = x2 - x1 + 1;
    int sy = y1 + (i * h) / OUTP;
    int ey = y1 + ((i + 1) * h + OUTP - 1) / OUTP;
    int sx = x1 + (j * w) / OUTP;
    int ex = x1 + ((j + 1) * w + OUTP - 1) / OUTP;
    sy = max(sy, 0); ey = min(ey, HH);
    sx = max(sx, 0); ex = min(ex, WW);

    const float* feat = x + ((size_t)roi_idx[r] * CC + c) * (HH * WW);
    float m = -3.402823466e+38f;
    for (int y = sy; y < ey; ++y) {
        const float* row = feat + y * WW;
        for (int xx = sx; xx < ex; ++xx) m = fmaxf(m, row[xx]);
    }
    outA[(size_t)r * DIN + c * (OUTP * OUTP) + ij] = (_Float16)m;
}

// ---------------------------------------------------------------------------
// 2) Split-K WMMA GEMM, double-buffered LDS, software-pipelined:
//      - A tile:  global_load_async_to_lds_b128 (ASYNCcnt, no VGPR roundtrip)
//      - B tile:  global f32 -> regs (issued early) -> cvt f16 -> LDS (after WMMA)
//    partial[s] = A(128 x K-slice) x B(K x N)
//    grid = (ceil(N/128), NSPLIT), block = 256 threads (8 wave32)
// ---------------------------------------------------------------------------
template <bool FAST>
__global__ void __launch_bounds__(256)
gemm128_wmma_f16(const _Float16* __restrict__ A,       // 128 x K row-major (f16)
                 const float*    __restrict__ B,       // K x N row-major (f32)
                 float*          __restrict__ partial, // NSPLIT x 128 x N
                 int N, int K)
{
    __shared__ _Float16 lA[2][BM][LDP];   // 24 KB
    __shared__ _Float16 lB[2][BN][LDP];   // 24 KB  (stored transposed: [n][k])

    const int tid   = threadIdx.x;
    const int lane  = tid & 31;
    const int wave  = tid >> 5;
    const int wm    = wave & 3;     // 4 wave-rows  -> 32 M each
    const int wn    = wave >> 2;    // 2 wave-cols  -> 64 N each
    const int lhalf = lane >> 4;    // 0: lanes 0-15, 1: lanes 16-31
    const int lidx  = lane & 15;

    const int n0 = blockIdx.x * BN;
    const int s  = blockIdx.y;
    const int Ks = K / NSPLIT;      // divisible by BK for all layers
    const int k0 = s * Ks;

    v8f acc[2][4];
#pragma unroll
    for (int mi = 0; mi < 2; ++mi)
#pragma unroll
        for (int ni = 0; ni < 4; ++ni)
            acc[mi][ni] = (v8f)(0.0f);

    // A staging: 128 rows, 2 threads/row, 32 bytes each (2 x b128 async)
    const int arow = tid >> 1;
    const int akc  = (tid & 1) * 16;
    // B staging (FAST): 4 k-rows x 4 n-cols per thread
    const int kq = (tid >> 5) * 4;   // k-quad base: 0,4,...,28
    const int nc = (tid & 31) * 4;   // n base: 0,4,...,124
    // B staging (slow): 1 k-row x 16 n per thread, per-element bounds
    const int bkr = tid >> 3;
    const int bnc = (tid & 7) * 16;

    float breg[4][4];

    // async copy of this thread's 32B slice of the A tile straight into LDS.
    // INST_OFFSET is added to BOTH the LDS and global address (ISA 08, §4.4),
    // so one (lds, global) pair covers both 16B halves.
    auto asyncA = [&](int kbase, int buf) {
        unsigned lds = (unsigned)(uintptr_t)&lA[buf][arow][akc];
        unsigned long long g =
            (unsigned long long)(uintptr_t)(A + (size_t)arow * K + kbase + akc);
        asm volatile("global_load_async_to_lds_b128 %0, %1, off\n\t"
                     "global_load_async_to_lds_b128 %0, %1, off offset:16"
                     :: "v"(lds), "v"(g) : "memory");
    };
    auto loadB = [&](int kbase) {
        if (FAST) {
            const float* b0 = B + (size_t)(kbase + kq) * N + n0 + nc;
#pragma unroll
            for (int r = 0; r < 4; ++r) {
                float4 q = *(const float4*)(b0 + (size_t)r * N);
                breg[r][0] = q.x; breg[r][1] = q.y; breg[r][2] = q.z; breg[r][3] = q.w;
            }
        } else {
#pragma unroll
            for (int u = 0; u < 16; ++u) {
                int gn = n0 + bnc + u;
                breg[u >> 2][u & 3] =
                    (gn < N) ? B[(size_t)(kbase + bkr) * N + gn] : 0.0f;
            }
        }
    };
    auto storeB = [&](int buf) {
        if (FAST) {
#pragma unroll
            for (int u = 0; u < 4; ++u) {
                v4h p = { (_Float16)breg[0][u], (_Float16)breg[1][u],
                          (_Float16)breg[2][u], (_Float16)breg[3][u] };
                *(v4h*)&lB[buf][nc + u][kq] = p;   // 8B-aligned
            }
        } else {
#pragma unroll
            for (int u = 0; u < 16; ++u)
                lB[buf][bnc + u][bkr] = (_Float16)breg[u >> 2][u & 3];
        }
    };
    // fragments per ISA 16-bit layout: lanes0-15 K=0-7/16-23, lanes16-31 K=8-15/24-31
    auto compute = [&](int buf) {
        v16h af[2], bf[4];
#pragma unroll
        for (int mi = 0; mi < 2; ++mi) {
            int row = wm * 32 + mi * 16 + lidx;
            v8h lo = *(const v8h*)&lA[buf][row][lhalf * 8];
            v8h hi = *(const v8h*)&lA[buf][row][16 + lhalf * 8];
#pragma unroll
            for (int e = 0; e < 8; ++e) { af[mi][e] = lo[e]; af[mi][8 + e] = hi[e]; }
        }
#pragma unroll
        for (int ni = 0; ni < 4; ++ni) {
            int col = wn * 64 + ni * 16 + lidx;
            v8h lo = *(const v8h*)&lB[buf][col][lhalf * 8];
            v8h hi = *(const v8h*)&lB[buf][col][16 + lhalf * 8];
#pragma unroll
            for (int e = 0; e < 8; ++e) { bf[ni][e] = lo[e]; bf[ni][8 + e] = hi[e]; }
        }
#pragma unroll
        for (int mi = 0; mi < 2; ++mi)
#pragma unroll
            for (int ni = 0; ni < 4; ++ni)
                acc[mi][ni] = __builtin_amdgcn_wmma_f32_16x16x32_f16(
                    false, af[mi], false, bf[ni],
                    (short)0, acc[mi][ni], false, false);
    };

    const int nsteps = Ks / BK;

    // prologue: stage tile 0
    asyncA(k0, 0);
    loadB(k0);
    storeB(0);
    asm volatile("s_wait_asynccnt 0x0" ::: "memory");
    __syncthreads();

    // steady state: issue tile step+1 loads, compute tile step, then drain
    for (int step = 0; step < nsteps - 1; ++step) {
        const int buf = step & 1;
        asyncA(k0 + (step + 1) * BK, buf ^ 1);
        loadB(k0 + (step + 1) * BK);
        if (FAST && (step + 2 < nsteps) && ((tid & 7) == 0)) {
            const float* pf = B + (size_t)(k0 + (step + 2) * BK + kq) * N + n0 + nc;
#pragma unroll
            for (int r = 0; r < 4; ++r)
                __builtin_prefetch(pf + (size_t)r * N, 0, 1);
        }
        compute(buf);          // WMMAs hide the global-load latency issued above
        storeB(buf ^ 1);       // loadcnt wait lands here, after the WMMA block
        asm volatile("s_wait_asynccnt 0x0" ::: "memory");
        __syncthreads();
    }
    compute((nsteps - 1) & 1); // epilogue tile

    // write this split's partial (C layout: vgpr r -> M = r + lhalf*8)
    float* pb = partial + (size_t)s * BM * N;
#pragma unroll
    for (int mi = 0; mi < 2; ++mi)
#pragma unroll
        for (int ni = 0; ni < 4; ++ni) {
            int gn = n0 + wn * 64 + ni * 16 + lidx;
            if (gn < N) {
#pragma unroll
                for (int r = 0; r < 8; ++r) {
                    int row = wm * 32 + mi * 16 + lhalf * 8 + r;
                    pb[(size_t)row * N + gn] = acc[mi][ni][r];
                }
            }
        }
}

// ---------------------------------------------------------------------------
// 3) Reduce splits + bias (+ReLU); output f16 (next layer) or f32 (final)
// ---------------------------------------------------------------------------
__global__ void reduce_epilogue(const float* __restrict__ partial,
                                const float* __restrict__ bias,
                                _Float16* __restrict__ out_h,
                                float*    __restrict__ out_f,
                                int N, int relu)
{
    int idx = blockIdx.x * blockDim.x + threadIdx.x;
    int total = BM * N;
    if (idx >= total) return;
    int n = idx % N;
    float v = 0.0f;
#pragma unroll
    for (int s = 0; s < NSPLIT; ++s)
        v += partial[(size_t)s * total + idx];
    v += bias[n];
    if (relu) v = fmaxf(v, 0.0f);
    if (out_h) out_h[idx] = (_Float16)v;
    else       out_f[idx] = v;
}

// ---------------------------------------------------------------------------
extern "C" void kernel_launch(void* const* d_in, const int* in_sizes, int n_in,
                              void* d_out, int out_size, void* d_ws, size_t ws_size,
                              hipStream_t stream)
{
    const float* x     = (const float*)d_in[0];
    const float* rois  = (const float*)d_in[1];
    const int*   ridx  = (const int*)  d_in[2];
    const float* w_fc6 = (const float*)d_in[3];
    const float* b_fc6 = (const float*)d_in[4];
    const float* w_fc7 = (const float*)d_in[5];
    const float* b_fc7 = (const float*)d_in[6];
    const float* w_cls = (const float*)d_in[7];
    const float* b_cls = (const float*)d_in[8];
    const float* w_sc  = (const float*)d_in[9];
    const float* b_sc  = (const float*)d_in[10];
    float* out = (float*)d_out;

    char* ws = (char*)d_ws;
    size_t off = 0;
    auto walloc = [&](size_t bytes) -> void* {
        void* p = ws + off;
        off = (off + bytes + 255) & ~(size_t)255;
        return p;
    };
    _Float16* A0   = (_Float16*)walloc((size_t)RR * DIN * sizeof(_Float16)); // 6.4 MB
    _Float16* A1   = (_Float16*)walloc((size_t)RR * DH  * sizeof(_Float16)); // 1 MB
    _Float16* A2   = (_Float16*)walloc((size_t)RR * DH  * sizeof(_Float16)); // 1 MB
    float*    part = (float*)   walloc((size_t)NSPLIT * RR * DH * sizeof(float)); // 16 MB

    // 1) RoI pool -> A0 (f16)
    {
        int tot = RR * CC * OUTP * OUTP;
        roi_pool_kernel<<<(tot + 255) / 256, 256, 0, stream>>>(x, rois, ridx, A0);
    }
    // 2) fc6 = relu(A0 @ w_fc6 + b)   (K=25088, N=4096) -> A1 f16
    gemm128_wmma_f16<true><<<dim3(DH / BN, NSPLIT), 256, 0, stream>>>(A0, w_fc6, part, DH, DIN);
    reduce_epilogue<<<(RR * DH + 255) / 256, 256, 0, stream>>>(part, b_fc6, A1, nullptr, DH, 1);
    // 3) fc7 = relu(A1 @ w_fc7 + b)   (K=4096, N=4096) -> A2 f16
    gemm128_wmma_f16<true><<<dim3(DH / BN, NSPLIT), 256, 0, stream>>>(A1, w_fc7, part, DH, DH);
    reduce_epilogue<<<(RR * DH + 255) / 256, 256, 0, stream>>>(part, b_fc7, A2, nullptr, DH, 1);
    // 4) roi_cls_locs = A2 @ w_cls + b  (N=84) -> d_out[0 : 128*84]
    gemm128_wmma_f16<false><<<dim3(1, NSPLIT), 256, 0, stream>>>(A2, w_cls, part, 84, DH);
    reduce_epilogue<<<(RR * 84 + 255) / 256, 256, 0, stream>>>(part, b_cls, nullptr, out, 84, 0);
    // 5) roi_scores = A2 @ w_score + b  (N=21) -> d_out[128*84 : ]
    gemm128_wmma_f16<false><<<dim3(1, NSPLIT), 256, 0, stream>>>(A2, w_sc, part, 21, DH);
    reduce_epilogue<<<(RR * 21 + 255) / 256, 256, 0, stream>>>(part, b_sc, nullptr, out + RR * 84, 21, 0);
}
